// QuantizedLinear_20469814133130
// MI455X (gfx1250) — compile-verified, matrix operand order checked by reading, not run on hardware
//
#include <hip/hip_runtime.h>

typedef _Float16 half2  __attribute__((ext_vector_type(2)));
typedef _Float16 half4  __attribute__((ext_vector_type(4)));
typedef _Float16 half8  __attribute__((ext_vector_type(8)));
typedef _Float16 half16 __attribute__((ext_vector_type(16)));
typedef float    float8 __attribute__((ext_vector_type(8)));
typedef float    f32x4  __attribute__((ext_vector_type(4)));
typedef int      i32x4  __attribute__((ext_vector_type(4)));
typedef unsigned u32x4  __attribute__((ext_vector_type(4)));

#define KDIM 8192   // IN_FEATURES
#define NDIM 8192   // OUT_FEATURES
#define BM   128    // batch rows per block
#define BN   256    // output cols per block
#define BK   32     // K per v_wmma_f32_16x16x32_f16 step
#define LDSS (BK + 8)          // padded LDS row stride in halves (80 B)
#define NT   (KDIM / BK)       // 256 K-steps

__device__ __forceinline__ void load_x_tile(const float* __restrict__ x, int mBase,
                                            int kt, int tid, f32x4 xr[4]) {
#pragma unroll
    for (int i = 0; i < 4; ++i) {
        const int f   = tid + i * 256;   // float4 id 0..1023 (128 rows x 8)
        const int row = f >> 3;
        const int c4  = f & 7;
        xr[i] = *(const f32x4*)(x + (mBase + row) * KDIM + kt + c4 * 4);
    }
}

__device__ __forceinline__ void store_x_tile(_Float16* __restrict__ xs, int tid,
                                             const f32x4 xr[4]) {
#pragma unroll
    for (int i = 0; i < 4; ++i) {
        const int f   = tid + i * 256;
        const int row = f >> 3;
        const int c4  = f & 7;
        half4 h;
        h[0] = (_Float16)xr[i][0]; h[1] = (_Float16)xr[i][1];
        h[2] = (_Float16)xr[i][2]; h[3] = (_Float16)xr[i][3];
        *(half4*)(&xs[row * LDSS + c4 * 4]) = h;
    }
}

__device__ __forceinline__ void load_w_tile(const int* __restrict__ pw, int nBase,
                                            int kt, int tid, i32x4 wr[4]) {
#pragma unroll
    for (int i = 0; i < 4; ++i) {
        const int f   = tid + i * 256;   // int4 id 0..1023 (256 rows x 4)
        const int row = f >> 2;
        const int c   = f & 3;
        wr[i] = *(const i32x4*)(pw + (nBase + row) * (KDIM / 2) + (kt >> 1) + c * 4);
    }
}

// decode one packed byte -> packed f16 pair {hi-8, lo-8} via 0x6400-bias trick
__device__ __forceinline__ unsigned dec2(int b) {
    const half2 bias = {(_Float16)1032.0f, (_Float16)1032.0f};  // 1024 + 8
    unsigned t = 0x64006400u | (unsigned)((b >> 4) & 15) | ((unsigned)(b & 15) << 16);
    half2 h = __builtin_bit_cast(half2, t) - bias;
    return __builtin_bit_cast(unsigned, h);
}

__device__ __forceinline__ void store_w_tile(_Float16* __restrict__ ws, int tid,
                                             const i32x4 wr[4]) {
#pragma unroll
    for (int i = 0; i < 4; ++i) {
        const int f   = tid + i * 256;
        const int row = f >> 2;
        const int c   = f & 3;
        u32x4 wv;
#pragma unroll
        for (int j = 0; j < 4; ++j) wv[j] = dec2(wr[i][j]);
        *(u32x4*)(&ws[row * LDSS + c * 8]) = wv;  // 8 halves, 16 B
    }
}

__device__ __forceinline__ half16 load_frag(const _Float16* __restrict__ p) {
    half8 lo = *(const half8*)p;
    half8 hi = *(const half8*)(p + 16);
    return __builtin_shufflevector(lo, hi, 0,1,2,3,4,5,6,7,8,9,10,11,12,13,14,15);
}

__global__ __launch_bounds__(256)
void q4_gemm_wmma(const float* __restrict__ x,
                  const int*   __restrict__ pw,
                  const float* __restrict__ scale,
                  float*       __restrict__ out)
{
    __shared__ _Float16 xs[2][BM * LDSS];   // 2 x 10 KB
    __shared__ _Float16 ws[2][BN * LDSS];   // 2 x 20 KB

    const int tid  = threadIdx.x;
    const int wave = tid >> 5;              // 0..7 (wave32)
    const int lane = tid & 31;

    const int mBase = blockIdx.y * BM;
    const int nBase = blockIdx.x * BN;

    // 2 waves along M (64 rows each), 4 along N (64 cols each): 4x4 fragments/wave
    const int wm = (wave & 1) * 64;
    const int wn = (wave >> 1) * 64;

    // WMMA 16-bit A/B fragment addressing: lane -> row/col (lane&15), K-base (lane>>4)*8
    const int lrow = lane & 15;
    const int lkb  = (lane >> 4) * 8;

    float8 acc[4][4];
#pragma unroll
    for (int mi = 0; mi < 4; ++mi)
#pragma unroll
        for (int ni = 0; ni < 4; ++ni)
            acc[mi][ni] = (float8)(0.0f);

    f32x4 xr[4];
    i32x4 wr[4];

    // prologue: stage tile 0 into buffer 0
    load_x_tile(x, mBase, 0, tid, xr);
    load_w_tile(pw, nBase, 0, tid, wr);
    store_x_tile(xs[0], tid, xr);
    store_w_tile(ws[0], tid, wr);

    for (int t = 0; t < NT; ++t) {
        const int cur = t & 1;

        // issue global loads for next tile (overlaps with compute below)
        if (t + 1 < NT) {
            load_x_tile(x, mBase, (t + 1) * BK, tid, xr);
            load_w_tile(pw, nBase, (t + 1) * BK, tid, wr);
        }

        __syncthreads();   // stores into buf[cur] complete; prev reads of buf[cur^1] done

        // ---- keep 4 A fragments live; stream B fragments one at a time ----
        half16 afrag[4];
#pragma unroll
        for (int mi = 0; mi < 4; ++mi)
            afrag[mi] = load_frag(&xs[cur][(wm + mi * 16 + lrow) * LDSS + lkb]);

#pragma unroll
        for (int ni = 0; ni < 4; ++ni) {
            half16 bfrag = load_frag(&ws[cur][(wn + ni * 16 + lrow) * LDSS + lkb]);
#pragma unroll
            for (int mi = 0; mi < 4; ++mi)
                acc[mi][ni] = __builtin_amdgcn_wmma_f32_16x16x32_f16(
                    false, afrag[mi], false, bfrag,
                    (short)0, acc[mi][ni], false, false);
        }

        // stage next tile into the other buffer (safe: barrier above fenced old readers)
        if (t + 1 < NT) {
            store_x_tile(xs[cur ^ 1], tid, xr);
            store_w_tile(ws[cur ^ 1], tid, wr);
        }
    }

    // ---- epilogue: y = acc * scale[col]; C/D layout: lane -> col (lane&15),
    //      vgpr j -> row j + (lane>>4)*8 ----
    const int rsel = (lane >> 4) * 8;
#pragma unroll
    for (int ni = 0; ni < 4; ++ni) {
        const int col = nBase + wn + ni * 16 + lrow;
        const float s = scale[col];
#pragma unroll
        for (int mi = 0; mi < 4; ++mi) {
            const int rbase = mBase + wm + mi * 16 + rsel;
#pragma unroll
            for (int j = 0; j < 8; ++j)
                out[(rbase + j) * NDIM + col] = acc[mi][ni][j] * s;
        }
    }
}

extern "C" void kernel_launch(void* const* d_in, const int* in_sizes, int n_in,
                              void* d_out, int out_size, void* d_ws, size_t ws_size,
                              hipStream_t stream) {
    const float* x     = (const float*)d_in[0];
    const int*   pw    = (const int*)d_in[1];
    const float* scale = (const float*)d_in[2];
    float*       out   = (float*)d_out;

    const int batch = in_sizes[0] / KDIM;            // 512
    dim3 grid(NDIM / BN, batch / BM);                // 32 x 4 blocks
    dim3 block(256);                                 // 8 wave32 waves
    q4_gemm_wmma<<<grid, block, 0, stream>>>(x, pw, scale, out);
}
